// SelfAttention_1494648619251
// MI455X (gfx1250) — compile-verified, matrix operand order checked by reading, not run on hardware
//
#include <hip/hip_runtime.h>

#define B_ 4
#define N_ 4096
#define D_ 64
#define BQ 128   // query rows per workgroup (8 waves x 16)
#define BK 32    // keys per inner iteration

typedef __bf16 bf16;
typedef __attribute__((ext_vector_type(16))) __bf16 v16bf;
typedef __attribute__((ext_vector_type(8)))  float  v8f;
typedef __attribute__((ext_vector_type(4)))  unsigned v4u;
typedef __attribute__((ext_vector_type(8)))  unsigned v8u;

union Frag {
  v16bf v;
  uint4 q[2];
  bf16  h[16];
};

// Pass 1: fp32 -> bf16, producing xb [B][N][D] and xbt [B][D][N] (transposed).
__global__ __launch_bounds__(256) void convert_kernel(const float* __restrict__ x,
                                                      bf16* __restrict__ xb,
                                                      bf16* __restrict__ xbt) {
  int idx = blockIdx.x * 256 + threadIdx.x;
  int e = idx * 4;                      // 4 contiguous elements along d
  float4 v = *(const float4*)(x + e);
  int b   = e >> 18;                    // / (N_*D_)
  int rem = e & (N_ * D_ - 1);
  int n   = rem >> 6;                   // / D_
  int d   = rem & (D_ - 1);
  union { bf16 h[4]; uint2 u; } t;
  t.h[0] = (bf16)v.x; t.h[1] = (bf16)v.y; t.h[2] = (bf16)v.z; t.h[3] = (bf16)v.w;
  *(uint2*)(xb + e) = t.u;
  #pragma unroll
  for (int i = 0; i < 4; ++i)
    xbt[((size_t)b * D_ + d + i) * N_ + n] = t.h[i];
}

// TDM: issue a 2D tensor_load_to_lds (bf16 elements) per ISA 08_async_tensor §8.
//   group0: count=1 | lds_addr | global_addr[56:0] | type=2
//   group1: data_size=2B | tensor_dim0/1 | tile_dim0/1 | tensor_dim0_stride
__device__ __attribute__((always_inline)) inline
void tdm_load_2d(unsigned lds_off, const void* gptr,
                 unsigned tensor_d0, unsigned tensor_d1,
                 unsigned tile_d0, unsigned tile_d1, unsigned stride0) {
  unsigned long long ga = (unsigned long long)(uintptr_t)gptr;
  v4u g0;
  g0[0] = 1u;                                   // count=1 (valid user descriptor)
  g0[1] = lds_off;                              // lds_addr (bytes)
  g0[2] = (unsigned)ga;                         // global_addr[31:0]
  g0[3] = (unsigned)(ga >> 32) | 0x80000000u;   // global_addr[56:32] | type=2<<30
  v8u g1;
  g1[0] = 0x00010000u;                          // data_size=1 (2 bytes); no pad/iter
  g1[1] = (tensor_d0 & 0xFFFFu) << 16;          // tensor_dim0[15:0] at bits 63:48
  g1[2] = (tensor_d0 >> 16) | ((tensor_d1 & 0xFFFFu) << 16);
  g1[3] = (tensor_d1 >> 16) | (tile_d0 << 16);  // tile_dim0 at bits 127:112
  g1[4] = tile_d1;                              // tile_dim1; tile_dim2=0 (2D)
  g1[5] = stride0;                              // tensor_dim0_stride[31:0]
  g1[6] = 0u;                                   // stride[47:32]=0; dim1_stride lo=0
  g1[7] = 0u;
  asm volatile("tensor_load_to_lds %0, %1" :: "s"(g0), "s"(g1) : "memory");
}

// Pass 2: flash attention. One wave owns 16 query rows.
// S^T = K x Q^T so softmax rows (queries) live on lanes; P^T in C-layout is
// bit-identical to the A-fragment layout needed for O += P x V.
// K/V tiles staged by the Tensor Data Mover, double-buffered against compute.
__global__ __launch_bounds__(256) void fa_kernel(const bf16* __restrict__ xb,
                                                 const bf16* __restrict__ xbt,
                                                 float* __restrict__ out) {
  const int qblk = blockIdx.x;
  const int b    = blockIdx.y;
  const int tid  = threadIdx.x;
  const int wave = tid >> 5;
  const int lane = tid & 31;
  const int lq   = lane & 15;   // query column / matrix N-lane
  const int hi   = lane >> 4;   // which half of the K-striding pattern

  __shared__ bf16 ldsK[2][BK * D_];  // key rows,  [32][64] per buffer
  __shared__ bf16 ldsV[2][D_ * BK];  // V^T rows,  [64][32] per buffer

  const bf16* xb_b  = xb  + (size_t)b * N_ * D_;
  const bf16* xbt_b = xbt + (size_t)b * D_ * N_;

  // Q as B-operand fragments (lane q holds Q[row q] with the K-striding).
  const int qrow = qblk * BQ + wave * 16 + lq;
  Frag qf[2];
  #pragma unroll
  for (int f = 0; f < 2; ++f) {
    const bf16* base = xb_b + (size_t)qrow * D_ + f * 32 + hi * 8;
    qf[f].q[0] = *(const uint4*)(base);       // elems 0..7  : d = f*32 + hi*8 ..
    qf[f].q[1] = *(const uint4*)(base + 16);  // elems 8..15 : d = f*32 + 16 + hi*8 ..
  }

  v8f o[4];
  #pragma unroll
  for (int dt = 0; dt < 4; ++dt) o[dt] = (v8f){0,0,0,0,0,0,0,0};

  float rmax = -__builtin_inff();
  float rsum = 0.0f;
  const float L2E = 1.4426950408889634f;

  // Prologue: wave 0 DMAs tile 0 into buffer 0.
  if (wave == 0) {
    tdm_load_2d((unsigned)(uintptr_t)&ldsK[0][0], xb_b, D_, N_, D_, BK, D_);
    tdm_load_2d((unsigned)(uintptr_t)&ldsV[0][0], xbt_b, N_, D_, BK, D_, N_);
  }

  for (int i = 0; i < N_ / BK; ++i) {
    const int j0  = i * BK;
    const int cur = i & 1;

    __syncthreads();  // (a) everyone finished reading buf[cur^1] last iter
    if (wave == 0) {
      if (j0 + BK < N_) {
        const int jn = j0 + BK;
        tdm_load_2d((unsigned)(uintptr_t)&ldsK[cur ^ 1][0],
                    xb_b + (size_t)jn * D_, D_, N_, D_, BK, D_);
        tdm_load_2d((unsigned)(uintptr_t)&ldsV[cur ^ 1][0],
                    xbt_b + jn, N_, D_, BK, D_, N_);
        __builtin_amdgcn_s_wait_tensorcnt(2);  // newest pair may fly; current done
      } else {
        __builtin_amdgcn_s_wait_tensorcnt(0);
      }
    }
    __syncthreads();  // (b) buf[cur] published to all waves

    // Burst-issue all fragment loads into independent registers so the
    // ds_load latency overlaps the WMMAs and softmax VALU (staggered dscnt).
    Frag ak[2][2];  // [key sub-tile][d chunk]
    #pragma unroll
    for (int sub = 0; sub < 2; ++sub) {
      #pragma unroll
      for (int f = 0; f < 2; ++f) {
        const bf16* base = &ldsK[cur][0] + (sub * 16 + lq) * D_ + f * 32 + hi * 8;
        ak[sub][f].q[0] = *(const uint4*)(base);
        ak[sub][f].q[1] = *(const uint4*)(base + 16);
      }
    }
    Frag bv[4];     // V B-fragments per 16-wide d tile (needed after softmax)
    #pragma unroll
    for (int dt = 0; dt < 4; ++dt) {
      const bf16* base = &ldsV[cur][0] + (dt * 16 + lq) * BK + hi * 8;
      bv[dt].q[0] = *(const uint4*)(base);       // keys hi*8..+7
      bv[dt].q[1] = *(const uint4*)(base + 16);  // keys 16+hi*8..+7
    }

    // S^T tiles: c[sub] = K[j0+sub*16 .. +16) x Q^T  (accumulate over d=64)
    v8f c[2];
    #pragma unroll
    for (int sub = 0; sub < 2; ++sub) {
      v8f acc = (v8f){0,0,0,0,0,0,0,0};
      #pragma unroll
      for (int f = 0; f < 2; ++f)
        acc = __builtin_amdgcn_wmma_f32_16x16x32_bf16(false, ak[sub][f].v, false,
                                                      qf[f].v, (short)0, acc,
                                                      false, false);
      c[sub] = acc;
    }

    // Online softmax: query lq's 32 scores are split lane/lane^16.
    float m_new = rmax;
    #pragma unroll
    for (int r = 0; r < 8; ++r) {
      m_new = fmaxf(m_new, c[0][r]);
      m_new = fmaxf(m_new, c[1][r]);
    }
    m_new = fmaxf(m_new, __shfl_xor(m_new, 16));
    const float scale = __builtin_amdgcn_exp2f((rmax - m_new) * L2E);
    rmax = m_new;

    // exp + pack: P^T C-layout == A-fragment layout for the PV matmul.
    Frag pa;
    float psum = 0.0f;
    #pragma unroll
    for (int r = 0; r < 8; ++r) {
      float p0 = __builtin_amdgcn_exp2f((c[0][r] - m_new) * L2E);  // keys (hi?8:0)+r
      float p1 = __builtin_amdgcn_exp2f((c[1][r] - m_new) * L2E);  // keys 16+..
      psum += p0 + p1;
      pa.h[r]     = (bf16)p0;
      pa.h[8 + r] = (bf16)p1;
    }
    rsum = rsum * scale + psum + __shfl_xor(psum, 16);

    // Rescale O: queries are the M (VGPR) dim there -> broadcast per-row scale.
    float sc[8];
    #pragma unroll
    for (int r = 0; r < 8; ++r) sc[r] = __shfl(scale, (hi << 3) + r, 32);
    #pragma unroll
    for (int dt = 0; dt < 4; ++dt) {
      #pragma unroll
      for (int r = 0; r < 8; ++r) o[dt][r] *= sc[r];
    }

    // O += P x V.
    #pragma unroll
    for (int dt = 0; dt < 4; ++dt)
      o[dt] = __builtin_amdgcn_wmma_f32_16x16x32_bf16(false, pa.v, false, bv[dt].v,
                                                      (short)0, o[dt], false, false);
  }

  // Normalize and store (lane holds column d = dt*16+lq for rows hi*8+r).
  float rs[8];
  #pragma unroll
  for (int r = 0; r < 8; ++r) rs[r] = 1.0f / __shfl(rsum, (hi << 3) + r, 32);

  const int qbase = qblk * BQ + wave * 16 + hi * 8;
  #pragma unroll
  for (int r = 0; r < 8; ++r) {
    float* orow = out + ((size_t)b * N_ + qbase + r) * D_;
    #pragma unroll
    for (int dt = 0; dt < 4; ++dt)
      orow[dt * 16 + lq] = o[dt][r] * rs[r];
  }
}

extern "C" void kernel_launch(void* const* d_in, const int* in_sizes, int n_in,
                              void* d_out, int out_size, void* d_ws, size_t ws_size,
                              hipStream_t stream) {
  const float* x = (const float*)d_in[0];
  float* out = (float*)d_out;
  bf16* xb  = (bf16*)d_ws;                        // 2 MB
  bf16* xbt = xb + (size_t)B_ * N_ * D_;          // 2 MB
  (void)in_sizes; (void)n_in; (void)out_size; (void)ws_size;

  convert_kernel<<<(B_ * N_ * D_) / (256 * 4), 256, 0, stream>>>(x, xb, xbt);
  dim3 grid(N_ / BQ, B_);
  fa_kernel<<<grid, 256, 0, stream>>>(xb, xbt, out);
}
